// Channel_Attention_18726057411037
// MI455X (gfx1250) — matrix operand order repaired and checked by reading, not compiled, not run on hardware
//
#include <hip/hip_runtime.h>
#include <hip/hip_bf16.h>
#include <math.h>

typedef __attribute__((ext_vector_type(16))) _Float16 v16h;
typedef __attribute__((ext_vector_type(8)))  _Float16 v8h;
typedef __attribute__((ext_vector_type(4)))  _Float16 v4h;
typedef __attribute__((ext_vector_type(8)))  float    v8f;
typedef __attribute__((ext_vector_type(4)))  float    v4f;

#define C_DIM   192
#define HEADS   8
#define HD      24
#define NTOK    16384   // 128*128
#define BATCH   16
#define QKV_R   576     // rows: [q(192), k(192), v(192)], head-major inside
#define KSPLIT  8       // K2 K-dimension split for occupancy

union FragU { v16h v; v8h h[2]; };

// A operand fragment (16x32 f16): lane = row l%16; one contiguous 16-half run at
// K offset (l/16)*16.  'p' already includes that offset.
static __device__ __forceinline__ v16h load_frag_a(const _Float16* p) {
  FragU f;
  f.h[0] = *(const v8h*)(p);
  f.h[1] = *(const v8h*)(p + 8);
  return f.v;
}
// B operand fragment (32x16 f16): lane = col l%16; one contiguous 16-half run at
// K offset (l/16)*16.  'p' already includes that offset.
static __device__ __forceinline__ v16h load_frag_b(const _Float16* p) {
  FragU f;
  f.h[0] = *(const v8h*)(p);
  f.h[1] = *(const v8h*)(p + 8);
  return f.v;
}

static __device__ __forceinline__ v8f wmma_f16(v16h a, v16h b, v8f c) {
  return __builtin_amdgcn_wmma_f32_16x16x32_f16(false, a, false, b, (short)0, c,
                                                false, false);
}

// one v_cvt_pk_rtz_f16_f32: convert+pack two f32 -> packed f16 pair
static __device__ __forceinline__ unsigned pack2f(float a, float b) {
  auto p = __builtin_amdgcn_cvt_pkrtz(a, b);
  return __builtin_bit_cast(unsigned, p);
}
// one v_perm_b32 each: interleave low/high halves of two half2 dwords
static __device__ __forceinline__ unsigned ilv_lo(unsigned a, unsigned b) {
  return __builtin_amdgcn_perm(b, a, 0x05040100u);  // {b.h0, a.h0}
}
static __device__ __forceinline__ unsigned ilv_hi(unsigned a, unsigned b) {
  return __builtin_amdgcn_perm(b, a, 0x07060302u);  // {b.h1, a.h1}
}

// ---------------------------------------------------------------------------
// K0: w_qkv [192][576] f32  ->  wT [576][192] f16   (row-major, K contiguous)
// ---------------------------------------------------------------------------
__global__ __launch_bounds__(256) void k0_wt(const float* __restrict__ w_qkv,
                                             _Float16* __restrict__ wT) {
  int idx = blockIdx.x * 256 + threadIdx.x;
  if (idx < QKV_R * C_DIM) {
    int r = idx / C_DIM, c = idx % C_DIM;
    wT[r * C_DIM + c] = (_Float16)w_qkv[c * QKV_R + r];
  }
}

// ---- K1 staging: thread owns 2 (c-pair, n4) items of the 32(K)x128(N) tile;
// rows 2cp,2cp+1 pack into one b32 LDS store at lx[n][2cp..2cp+1].
static __device__ __forceinline__ void k1_stage_load(const float* __restrict__ xb,
                                                     int ks, int tid, v4f f[4]) {
  for (int it = 0; it < 2; ++it) {
    int j = tid + it * 256;            // 0..511
    int cp = j >> 5, n4 = j & 31;
    const float* base = xb + (size_t)(ks * 32 + 2 * cp) * NTOK + n4 * 4;
    f[2 * it]     = *(const v4f*)(base);
    f[2 * it + 1] = *(const v4f*)(base + NTOK);
  }
}
static __device__ __forceinline__ void k1_stage_store(unsigned* __restrict__ lw,
                                                      int tid, const v4f f[4]) {
  for (int it = 0; it < 2; ++it) {
    int j = tid + it * 256;
    int cp = j >> 5, n4 = j & 31;
    for (int i = 0; i < 4; ++i)
      lw[(n4 * 4 + i) * 16 + cp] = pack2f(f[2 * it][i], f[2 * it + 1][i]);
  }
}

// ---------------------------------------------------------------------------
// K1: qkv[b][row 0..575][n] = sum_c wT[row][c] * x[b][c][n]   (f16 out)
// + sum-of-squares over n for q/k rows into norms (for fused L2 normalize).
// 8 waves as 4(M)x2(N); block tile M=64,N=128,K=192; double-buffered LDS.
// ---------------------------------------------------------------------------
__global__ __launch_bounds__(256) void k1_qkv(const float* __restrict__ x,
                                              const _Float16* __restrict__ wT,
                                              _Float16* __restrict__ qkv,
                                              float* __restrict__ norms) {
  __shared__ _Float16 lx[2][128 * 32];
  __shared__ float s_norm[64];

  const int tid  = threadIdx.x;
  const int lane = tid & 31;
  const int wv   = tid >> 5;
  const int wm   = wv & 3;
  const int wn   = wv >> 2;
  const int lr   = lane & 15;
  const int hi   = lane >> 4;
  const int nblock = blockIdx.x * 128;
  const int mblock = blockIdx.y * 64;
  const int b      = blockIdx.z;

  if (tid < 64) s_norm[tid] = 0.0f;

  const float* xb = x + (size_t)b * C_DIM * NTOK + nblock;

  v8f vz = {};
  v8f acc[4];
  for (int i = 0; i < 4; ++i) acc[i] = vz;

  v4f f[4];
  k1_stage_load(xb, 0, tid, f);
  k1_stage_store((unsigned*)&lx[0][0], tid, f);

  int p = 0;
  for (int ks = 0; ks < 6; ++ks) {
    if (ks < 5) k1_stage_load(xb, ks + 1, tid, f);  // in flight during WMMAs
    __syncthreads();

    v16h a = load_frag_a(wT + (size_t)(mblock + wm * 16 + lr) * C_DIM +
                         ks * 32 + hi * 16);
    const _Float16* lb = &lx[p][0] + (wn * 64 + lr) * 32 + hi * 16;
    for (int nt = 0; nt < 4; ++nt)
      acc[nt] = wmma_f16(a, load_frag_b(lb + nt * 16 * 32), acc[nt]);

    if (ks < 5) {
      k1_stage_store((unsigned*)&lx[1 - p][0], tid, f);
      p ^= 1;
    }
  }
  __syncthreads();

  const bool qk = (mblock < 384);
  for (int nt = 0; nt < 4; ++nt) {
    int gcol = nblock + wn * 64 + nt * 16 + lr;
    for (int r = 0; r < 8; ++r) {
      int lrow = wm * 16 + r + hi * 8;
      float v = acc[nt][r];
      qkv[((size_t)b * QKV_R + mblock + lrow) * NTOK + gcol] = (_Float16)v;
      if (qk) atomicAdd(&s_norm[lrow], v * v);
    }
  }
  __syncthreads();
  if (qk && tid < 64) atomicAdd(&norms[b * 384 + mblock + tid], s_norm[tid]);
}

// ---------------------------------------------------------------------------
// K2: partial raw scores S += Q Kt over a K-slice (WMMA, f32 acc), LDS-reduce
// across the block's 4 waves, then global f32 atomicAdd into sraw[b][h][24][24].
// Tile rows/cols 24..31 read harmless in-bounds garbage (never accumulated).
// Grid (HEADS, BATCH, KSPLIT) for occupancy: 4096 waves streaming q/k.
// ---------------------------------------------------------------------------
__global__ __launch_bounds__(128) void k2_attn(const _Float16* __restrict__ qkv,
                                               float* __restrict__ sraw) {
  __shared__ float sS[32][32];
  const int tid  = threadIdx.x;
  const int lane = tid & 31;
  const int wv   = tid >> 5;
  const int lr   = lane & 15;
  const int hi   = lane >> 4;
  const int h = blockIdx.x, b = blockIdx.y, kz = blockIdx.z;

  for (int i = tid; i < 32 * 32; i += 128) (&sS[0][0])[i] = 0.0f;

  const size_t qbase = ((size_t)b * QKV_R + h * HD) * NTOK;
  const size_t kbase = ((size_t)b * QKV_R + 192 + h * HD) * NTOK;

  v8f vz = {};
  v8f acc[2][2];
  acc[0][0] = vz; acc[0][1] = vz; acc[1][0] = vz; acc[1][1] = vz;

  const int k0 = (kz * 4 + wv) * (NTOK / (4 * KSPLIT));
#pragma unroll 2
  for (int k = k0; k < k0 + NTOK / (4 * KSPLIT); k += 32) {
    v16h a0 = load_frag_a(qkv + qbase + (size_t)(lr)      * NTOK + k + hi * 16);
    v16h a1 = load_frag_a(qkv + qbase + (size_t)(16 + lr) * NTOK + k + hi * 16);
    v16h b0 = load_frag_b(qkv + kbase + (size_t)(lr)      * NTOK + k + hi * 16);
    v16h b1 = load_frag_b(qkv + kbase + (size_t)(16 + lr) * NTOK + k + hi * 16);
    acc[0][0] = wmma_f16(a0, b0, acc[0][0]);
    acc[0][1] = wmma_f16(a0, b1, acc[0][1]);
    acc[1][0] = wmma_f16(a1, b0, acc[1][0]);
    acc[1][1] = wmma_f16(a1, b1, acc[1][1]);
  }
  __syncthreads();
  for (int mt = 0; mt < 2; ++mt)
    for (int nt = 0; nt < 2; ++nt)
      for (int r = 0; r < 8; ++r)
        atomicAdd(&sS[mt * 16 + r + hi * 8][nt * 16 + lr], acc[mt][nt][r]);
  __syncthreads();

  float* sp = sraw + ((size_t)b * HEADS + h) * HD * HD;
  for (int idx = tid; idx < HD * HD; idx += 128) {
    int i = idx / HD, j = idx % HD;
    atomicAdd(&sp[idx], sS[i][j]);
  }
}

// ---------------------------------------------------------------------------
// K3: fold norms + temperature + softmax over raw scores (into LDS), then
// M_b = w_projT @ blockdiag(attn_b)  ->  f16 [b][192][192], K contiguous.
// ---------------------------------------------------------------------------
__global__ __launch_bounds__(256) void k3_mix(const float* __restrict__ w_proj,
                                              const float* __restrict__ sraw,
                                              const float* __restrict__ norms,
                                              const float* __restrict__ temp,
                                              _Float16* __restrict__ Mm) {
  __shared__ float sa[HEADS * HD * HD];   // post-softmax attention, 18 KB
  const int b = blockIdx.x;
  const int tid = threadIdx.x;

  if (tid < HEADS * HD) {                 // one (head, row) per thread
    int h = tid / HD, i = tid % HD;
    float nq = fmaxf(sqrtf(norms[b * 384 + h * HD + i]), 1e-12f);
    float t  = temp[h];
    const float* sp = sraw + ((size_t)b * HEADS + h) * HD * HD + i * HD;
    float row[HD];
    float m = -3.402823466e38f;
    for (int j = 0; j < HD; ++j) {
      float nk = fmaxf(sqrtf(norms[b * 384 + 192 + h * HD + j]), 1e-12f);
      float v = sp[j] / (nq * nk) * t;
      row[j] = v;
      m = fmaxf(m, v);
    }
    float s = 0.0f;
    for (int j = 0; j < HD; ++j) { row[j] = __expf(row[j] - m); s += row[j]; }
    float inv = 1.0f / s;
    for (int j = 0; j < HD; ++j) sa[(h * HD + i) * HD + j] = row[j] * inv;
  }
  __syncthreads();

  for (int t = tid; t < C_DIM * C_DIM; t += 256) {
    int i  = t / C_DIM;
    int cp = t % C_DIM;
    int h = cp / HD, e = cp % HD;
    float s = 0.0f;
    for (int dd = 0; dd < HD; ++dd)
      s += w_proj[(h * HD + dd) * C_DIM + i] * sa[(h * HD + dd) * HD + e];
    Mm[(size_t)b * C_DIM * C_DIM + i * C_DIM + cp] = (_Float16)s;
  }
}

// ---- K4 staging (f16 source): interleave two rows with v_perm_b32
static __device__ __forceinline__ void k4_stage_load(const _Float16* __restrict__ vb,
                                                     int ks, int tid, v4h f[4]) {
  for (int it = 0; it < 2; ++it) {
    int j = tid + it * 256;
    int cp = j >> 5, n4 = j & 31;
    const _Float16* base = vb + (size_t)(ks * 32 + 2 * cp) * NTOK + n4 * 4;
    f[2 * it]     = *(const v4h*)(base);
    f[2 * it + 1] = *(const v4h*)(base + NTOK);
  }
}
static __device__ __forceinline__ void k4_stage_store(unsigned* __restrict__ lw,
                                                      int tid, const v4h f[4]) {
  for (int it = 0; it < 2; ++it) {
    int j = tid + it * 256;
    int cp = j >> 5, n4 = j & 31;
    union { v4h h; unsigned u[2]; } a, bb;
    a.h  = f[2 * it];
    bb.h = f[2 * it + 1];
    lw[(n4 * 4 + 0) * 16 + cp] = ilv_lo(a.u[0], bb.u[0]);
    lw[(n4 * 4 + 1) * 16 + cp] = ilv_hi(a.u[0], bb.u[0]);
    lw[(n4 * 4 + 2) * 16 + cp] = ilv_lo(a.u[1], bb.u[1]);
    lw[(n4 * 4 + 3) * 16 + cp] = ilv_hi(a.u[1], bb.u[1]);
  }
}

// ---------------------------------------------------------------------------
// K4: out[b][i][n] = sum_c Mm[b][i][c] * v[b][c][n] + bias[i]   (f32 out)
// v rows are qkv rows 384..575.  Same double-buffered blocking as K1.
// ---------------------------------------------------------------------------
__global__ __launch_bounds__(256) void k4_out(const _Float16* __restrict__ qkv,
                                              const _Float16* __restrict__ Mm,
                                              const float* __restrict__ bias,
                                              float* __restrict__ out) {
  __shared__ _Float16 lx[2][128 * 32];

  const int tid  = threadIdx.x;
  const int lane = tid & 31;
  const int wv   = tid >> 5;
  const int wm   = wv & 3;
  const int wn   = wv >> 2;
  const int lr   = lane & 15;
  const int hi   = lane >> 4;
  const int nblock = blockIdx.x * 128;
  const int mblock = blockIdx.y * 64;
  const int b      = blockIdx.z;

  const _Float16* vb = qkv + ((size_t)b * QKV_R + 384) * NTOK + nblock;
  const _Float16* Ma = Mm + (size_t)b * C_DIM * C_DIM +
                       (mblock + wm * 16 + lr) * C_DIM;

  v8f vz = {};
  v8f acc[4];
  for (int i = 0; i < 4; ++i) acc[i] = vz;

  v4h f[4];
  k4_stage_load(vb, 0, tid, f);
  k4_stage_store((unsigned*)&lx[0][0], tid, f);

  int p = 0;
  for (int ks = 0; ks < 6; ++ks) {
    if (ks < 5) k4_stage_load(vb, ks + 1, tid, f);
    __syncthreads();

    v16h a = load_frag_a(Ma + ks * 32 + hi * 16);
    const _Float16* lb = &lx[p][0] + (wn * 64 + lr) * 32 + hi * 16;
    for (int nt = 0; nt < 4; ++nt)
      acc[nt] = wmma_f16(a, load_frag_b(lb + nt * 16 * 32), acc[nt]);

    if (ks < 5) {
      k4_stage_store((unsigned*)&lx[1 - p][0], tid, f);
      p ^= 1;
    }
  }

  for (int nt = 0; nt < 4; ++nt) {
    int gcol = nblock + wn * 64 + nt * 16 + lr;
    for (int r = 0; r < 8; ++r) {
      int grow = mblock + wm * 16 + r + hi * 8;
      out[((size_t)b * C_DIM + grow) * NTOK + gcol] = acc[nt][r] + bias[grow];
    }
  }
}

// ---------------------------------------------------------------------------
extern "C" void kernel_launch(void* const* d_in, const int* in_sizes, int n_in,
                              void* d_out, int out_size, void* d_ws,
                              size_t ws_size, hipStream_t stream) {
  const float* x     = (const float*)d_in[0];
  const float* w_qkv = (const float*)d_in[1];
  const float* w_prj = (const float*)d_in[2];
  const float* b_prj = (const float*)d_in[3];
  const float* temp  = (const float*)d_in[4];
  float* out = (float*)d_out;

  char* ws = (char*)d_ws;
  size_t o = 0;
  _Float16* qkv = (_Float16*)(ws + o); o += (size_t)BATCH * QKV_R * NTOK * 2;
  _Float16* wT  = (_Float16*)(ws + o); o += (size_t)QKV_R * C_DIM * 2;
  float*    nrm = (float*)(ws + o);    o += (size_t)BATCH * 384 * 4;
  float*    srw = (float*)(ws + o);    o += (size_t)BATCH * HEADS * HD * HD * 4;
  _Float16* Mm  = (_Float16*)(ws + o);

  hipMemsetAsync(nrm, 0, (size_t)BATCH * 384 * sizeof(float), stream);
  hipMemsetAsync(srw, 0, (size_t)BATCH * HEADS * HD * HD * sizeof(float), stream);
  k0_wt<<<(QKV_R * C_DIM + 255) / 256, 256, 0, stream>>>(w_qkv, wT);
  k1_qkv<<<dim3(NTOK / 128, QKV_R / 64, BATCH), 256, 0, stream>>>(x, wT, qkv, nrm);
  k2_attn<<<dim3(HEADS, BATCH, KSPLIT), 128, 0, stream>>>(qkv, srw);
  k3_mix<<<BATCH, 256, 0, stream>>>(w_prj, srw, nrm, temp, Mm);
  k4_out<<<dim3(NTOK / 128, C_DIM / 64, BATCH), 256, 0, stream>>>(qkv, Mm, b_prj, out);
}
